// CrossAttentionModel_73005854097706
// MI455X (gfx1250) — compile-verified, matrix operand order checked by reading, not run on hardware
//
#include <hip/hip_runtime.h>

#define D_DIM 1024
#define BATCH 64
#define SEQ   512

#define BM 128
#define BN 128
#define BK 32
#define KT (D_DIM / BK)   // 32 k-tiles
#define LDT 40   // padded LDS row stride (bf16 elems); 80 bytes/row keeps 16B alignment every row

typedef __attribute__((ext_vector_type(16))) __bf16 v16bf;
typedef __attribute__((ext_vector_type(8)))  float  v8f;

__device__ __forceinline__ __bf16 f2bf(float f) { return (__bf16)f; }
__device__ __forceinline__ float  bf2f(__bf16 b) { return (float)b; }

// Load a 16x32 bf16 fragment (A or B operand) from an LDS tile laid out
// row-major with stride LDT. Per ISA 16-bit A layout: lanes 0-15 hold rows
// 0-15 with K=0..7 (v0-3) and K=16..23 (v4-7); lanes 16-31 hold K=8..15 and
// K=24..31 of the same rows.
__device__ __forceinline__ v16bf load_frag(const __bf16* t, int row0) {
    int lane = threadIdx.x & 31;
    int r  = row0 + (lane & 15);
    int kh = (lane >> 4) << 3;          // 0 or 8
    const __bf16* p = t + r * LDT + kh;
    v16bf v;
#pragma unroll
    for (int i = 0; i < 8; ++i) v[i] = p[i];
#pragma unroll
    for (int i = 0; i < 8; ++i) v[8 + i] = p[16 + i];
    return v;
}

__device__ __forceinline__ v8f wmma_bf16(v16bf a, v16bf b, v8f c) {
    // (neg_a, A, neg_b, B, c_mod, C, reuse_a, reuse_b)
    return __builtin_amdgcn_wmma_f32_16x16x32_bf16(false, a, false, b, (short)0, c, false, false);
}

// One per-lane 16-byte async copy global -> LDS (GVS mode:
// addr = SADDR(64) + VADDR(32, bytes); dsaddr = LDS_BASE + VDST). ASYNCcnt.
__device__ __forceinline__ void async_copy_b128(unsigned lds_off, unsigned gbyte_off,
                                                const void* sbase) {
    asm volatile("global_load_async_to_lds_b128 %0, %1, %2"
                 :
                 : "v"(lds_off), "v"(gbyte_off), "s"((unsigned long long)sbase)
                 : "memory");
}
__device__ __forceinline__ void async_wait0() {
    asm volatile("s_wait_asynccnt 0x0" ::: "memory");
}
// Allow one prefetched tile (4 per-thread copies) to remain in flight.
__device__ __forceinline__ void async_wait4() {
    asm volatile("s_wait_asynccnt 0x4" ::: "memory");
}
__device__ __forceinline__ unsigned lds_off_of(const void* p) {
    // low 32 bits of a generic pointer into __shared__ = offset within the
    // workgroup's LDS allocation (hardware adds LDS_BASE).
    return (unsigned)(unsigned long long)p;
}

// ---------------------------------------------------------------------------
// Kernel 1: Y[m,n] = sum_k X[m,k] * W[n,k] + bias[n]   (fp32 in, bf16 out)
// Register-prefetch pipeline: next tile's global loads are in flight while
// the current tile's 8 WMMAs execute from LDS.
// Grid: (D/BN, M/BM). 8 waves, each owns 64(M) x 32(N) = 4x2 wmma tiles.
// ---------------------------------------------------------------------------
__global__ __launch_bounds__(256)
void proj_kernel(const float* __restrict__ X, const float* __restrict__ W,
                 const float* __restrict__ bias, __bf16* __restrict__ Y) {
    __shared__ alignas(16) __bf16 As[BM * LDT];
    __shared__ alignas(16) __bf16 Bs[BN * LDT];

    const int tid  = threadIdx.x;
    const int m0   = blockIdx.y * BM;
    const int n0   = blockIdx.x * BN;
    const int wave = tid >> 5;
    const int wm   = (wave >> 2) * 64;   // 0 or 64
    const int wn   = (wave & 3)  * 32;   // 0,32,64,96

    float4 xr[4], wr[4];
    auto gload = [&](int k0) {
#pragma unroll
        for (int it = 0; it < 4; ++it) {
            int idx = tid + it * 256;           // 0..1023, 8 float4 per row
            int r = idx >> 3;
            int c = (idx & 7) << 2;
            xr[it] = *(const float4*)(X + (size_t)(m0 + r) * D_DIM + k0 + c);
            wr[it] = *(const float4*)(W + (size_t)(n0 + r) * D_DIM + k0 + c);
        }
    };
    auto lstore = [&]() {
#pragma unroll
        for (int it = 0; it < 4; ++it) {
            int idx = tid + it * 256;
            int r = idx >> 3;
            int c = (idx & 7) << 2;
            __bf16* da = As + r * LDT + c;
            da[0] = f2bf(xr[it].x); da[1] = f2bf(xr[it].y);
            da[2] = f2bf(xr[it].z); da[3] = f2bf(xr[it].w);
            __bf16* db = Bs + r * LDT + c;
            db[0] = f2bf(wr[it].x); db[1] = f2bf(wr[it].y);
            db[2] = f2bf(wr[it].z); db[3] = f2bf(wr[it].w);
        }
    };

    v8f acc[4][2];
#pragma unroll
    for (int i = 0; i < 4; ++i)
#pragma unroll
        for (int j = 0; j < 2; ++j)
#pragma unroll
            for (int e = 0; e < 8; ++e) acc[i][j][e] = 0.0f;

    gload(0);
    lstore();
    __syncthreads();

    for (int kt = 0; kt < KT; ++kt) {
        if (kt + 1 < KT) gload((kt + 1) * BK);   // overlap with WMMAs below

        v16bf a[4], b[2];
#pragma unroll
        for (int i = 0; i < 4; ++i) a[i] = load_frag(As, wm + 16 * i);
#pragma unroll
        for (int j = 0; j < 2; ++j) b[j] = load_frag(Bs, wn + 16 * j);
#pragma unroll
        for (int i = 0; i < 4; ++i)
#pragma unroll
            for (int j = 0; j < 2; ++j)
                acc[i][j] = wmma_bf16(a[i], b[j], acc[i][j]);

        if (kt + 1 < KT) {
            __syncthreads();        // all waves done reading the LDS tile
            lstore();               // convert + store prefetched tile
            __syncthreads();
        }
    }

    // epilogue: C layout — VGPR v, lanes 0-15: M=v,N=lane; lanes 16-31: M=v+8,N=lane-16
    const int lane = tid & 31;
    const int nn = lane & 15;
    const int mb = (lane >> 4) << 3;
#pragma unroll
    for (int i = 0; i < 4; ++i)
#pragma unroll
        for (int j = 0; j < 2; ++j) {
            int gn = n0 + wn + 16 * j + nn;
            int gm = m0 + wm + 16 * i + mb;
            float bv = bias[gn];
#pragma unroll
            for (int v = 0; v < 8; ++v)
                Y[(size_t)(gm + v) * D_DIM + gn] = f2bf(acc[i][j][v] + bv);
        }
}

// ---------------------------------------------------------------------------
// Kernel 2: S[b] = Q[b] (SEQ x D) @ K[b]^T (D x SEQ), bf16 in, fp32 out.
// Double-buffered GLOBAL_LOAD_ASYNC_TO_LDS_B128 pipeline: tile k+1 streams
// into the other LDS buffer while tile k's WMMAs run; s_wait_asynccnt 4
// (async loads complete in order; 4 copies/thread/tile) gates the current
// tile only.
// Grid: (SEQ/BN, SEQ/BM, BATCH)
// ---------------------------------------------------------------------------
__global__ __launch_bounds__(256)
void scores_kernel(const __bf16* __restrict__ Q, const __bf16* __restrict__ K,
                   float* __restrict__ S) {
    __shared__ alignas(16) __bf16 As[2][BM * LDT];
    __shared__ alignas(16) __bf16 Bs[2][BN * LDT];

    const int tid  = threadIdx.x;
    const int bat  = blockIdx.z;
    const __bf16* Qb = Q + (size_t)bat * SEQ * D_DIM;
    const __bf16* Kb = K + (size_t)bat * SEQ * D_DIM;
    float* Sb = S + (size_t)bat * SEQ * SEQ;

    const int m0   = blockIdx.y * BM;
    const int n0   = blockIdx.x * BN;
    const int wave = tid >> 5;
    const int wm   = (wave >> 2) * 64;
    const int wn   = (wave & 3)  * 32;

    auto issue_tile = [&](int k0, int buf) {
#pragma unroll
        for (int it = 0; it < 2; ++it) {
            int idx = tid + it * 256;           // 0..511, 4 x 16B chunks per row
            int r = idx >> 2;
            int c = (idx & 3) << 3;
            unsigned ldsA = lds_off_of(&As[buf][r * LDT + c]);
            unsigned ldsB = lds_off_of(&Bs[buf][r * LDT + c]);
            unsigned offA = (unsigned)(((m0 + r) * D_DIM + k0 + c) * (int)sizeof(__bf16));
            unsigned offB = (unsigned)(((n0 + r) * D_DIM + k0 + c) * (int)sizeof(__bf16));
            async_copy_b128(ldsA, offA, Qb);
            async_copy_b128(ldsB, offB, Kb);
        }
    };

    v8f acc[4][2];
#pragma unroll
    for (int i = 0; i < 4; ++i)
#pragma unroll
        for (int j = 0; j < 2; ++j)
#pragma unroll
            for (int e = 0; e < 8; ++e) acc[i][j][e] = 0.0f;

    issue_tile(0, 0);

    for (int kt = 0; kt < KT; ++kt) {
        const int cur = kt & 1;
        if (kt + 1 < KT) {
            issue_tile((kt + 1) * BK, cur ^ 1);
            async_wait4();          // current tile resident; next may be in flight
        } else {
            async_wait0();
        }
        __syncthreads();

        v16bf a[4], b[2];
#pragma unroll
        for (int i = 0; i < 4; ++i) a[i] = load_frag(As[cur], wm + 16 * i);
#pragma unroll
        for (int j = 0; j < 2; ++j) b[j] = load_frag(Bs[cur], wn + 16 * j);
#pragma unroll
        for (int i = 0; i < 4; ++i)
#pragma unroll
            for (int j = 0; j < 2; ++j)
                acc[i][j] = wmma_bf16(a[i], b[j], acc[i][j]);
        __syncthreads();            // protect buf reuse by tile kt+2's issue
    }

    const int lane = tid & 31;
    const int nn = lane & 15;
    const int mb = (lane >> 4) << 3;
#pragma unroll
    for (int i = 0; i < 4; ++i)
#pragma unroll
        for (int j = 0; j < 2; ++j) {
            int gn = n0 + wn + 16 * j + nn;
            int gm = m0 + wm + 16 * i + mb;
#pragma unroll
            for (int v = 0; v < 8; ++v)
                Sb[(size_t)(gm + v) * SEQ + gn] = acc[i][j][v];
        }
}

// ---------------------------------------------------------------------------
// Kernel 3: per-row softmax stats. One wave per row of S (B*SEQ rows).
// ---------------------------------------------------------------------------
__global__ __launch_bounds__(256)
void row_stats_kernel(const float* __restrict__ S, float* __restrict__ Mx,
                      float* __restrict__ Rz) {
    int row  = blockIdx.x * 8 + (threadIdx.x >> 5);
    int lane = threadIdx.x & 31;
    const float* p = S + (size_t)row * SEQ;
    float m = -3.4e38f;
    for (int c = lane; c < SEQ; c += 32) m = fmaxf(m, p[c]);
#pragma unroll
    for (int off = 16; off > 0; off >>= 1) m = fmaxf(m, __shfl_xor(m, off, 32));
    float z = 0.0f;
    for (int c = lane; c < SEQ; c += 32) z += __expf(p[c] - m);
#pragma unroll
    for (int off = 16; off > 0; off >>= 1) z += __shfl_xor(z, off, 32);
    if (lane == 0) { Mx[row] = m; Rz[row] = 1.0f / z; }
}

// ---------------------------------------------------------------------------
// Kernel 4: w[b,k] = sum_q exp(S[b,q,k]-m[b,q]) * rz[b,q]   (coalesced in k)
// Grid: (SEQ/256, BATCH)
// ---------------------------------------------------------------------------
__global__ __launch_bounds__(256)
void colsum_kernel(const float* __restrict__ S, const float* __restrict__ Mx,
                   const float* __restrict__ Rz, float* __restrict__ Wv) {
    int b = blockIdx.y;
    int k = blockIdx.x * 256 + threadIdx.x;
    const float* Sb = S + (size_t)b * SEQ * SEQ;
    const float* mb = Mx + b * SEQ;
    const float* rb = Rz + b * SEQ;
    float acc = 0.0f;
    for (int q = 0; q < SEQ; ++q)
        acc += __expf(Sb[(size_t)q * SEQ + k] - mb[q]) * rb[q];
    Wv[b * SEQ + k] = acc;
}

// ---------------------------------------------------------------------------
// Kernel 5: out[b,d] = sum_k w[b,k] * V2[b,k,d]
// Grid: (D/256, BATCH)
// ---------------------------------------------------------------------------
__global__ __launch_bounds__(256)
void out_kernel(const float* __restrict__ Wv, const __bf16* __restrict__ V2,
                float* __restrict__ Out) {
    int b = blockIdx.y;
    int d = blockIdx.x * 256 + threadIdx.x;
    const __bf16* Vb = V2 + (size_t)b * SEQ * D_DIM;
    const float*  wb = Wv + b * SEQ;
    float acc = 0.0f;
    for (int k = 0; k < SEQ; ++k)
        acc += wb[k] * bf2f(Vb[(size_t)k * D_DIM + d]);
    Out[b * D_DIM + d] = acc;
}

extern "C" void kernel_launch(void* const* d_in, const int* in_sizes, int n_in,
                              void* d_out, int out_size, void* d_ws, size_t ws_size,
                              hipStream_t stream) {
    (void)in_sizes; (void)n_in; (void)out_size; (void)ws_size;
    const float* input1 = (const float*)d_in[0];
    const float* input2 = (const float*)d_in[1];
    const float* W_q1   = (const float*)d_in[2];
    const float* b_q1   = (const float*)d_in[3];
    const float* W_k2   = (const float*)d_in[10];
    const float* b_k2   = (const float*)d_in[11];
    const float* W_v2   = (const float*)d_in[12];
    const float* b_v2   = (const float*)d_in[13];
    float* out = (float*)d_out;

    // workspace layout
    char* ws = (char*)d_ws;
    const size_t szBF = (size_t)BATCH * SEQ * D_DIM * sizeof(__bf16); // 64 MB
    const size_t szSC = (size_t)BATCH * SEQ * SEQ * sizeof(float);    // 64 MB
    const size_t szV  = (size_t)BATCH * SEQ * sizeof(float);          // 128 KB
    __bf16* Q1 = (__bf16*)(ws);
    __bf16* K2 = (__bf16*)(ws + szBF);
    __bf16* V2 = (__bf16*)(ws + 2 * szBF);
    float*  Sc = (float*) (ws + 3 * szBF);
    float*  Mx = (float*) (ws + 3 * szBF + szSC);
    float*  Rz = (float*) (ws + 3 * szBF + szSC + szV);
    float*  Wv = (float*) (ws + 3 * szBF + szSC + 2 * szV);

    dim3 blk(256);
    dim3 gProj(D_DIM / BN, (BATCH * SEQ) / BM);          // (8, 256)
    proj_kernel<<<gProj, blk, 0, stream>>>(input1, W_q1, b_q1, Q1);
    proj_kernel<<<gProj, blk, 0, stream>>>(input2, W_k2, b_k2, K2);
    proj_kernel<<<gProj, blk, 0, stream>>>(input2, W_v2, b_v2, V2);

    dim3 gS(SEQ / BN, SEQ / BM, BATCH);                  // (4, 4, 64)
    scores_kernel<<<gS, blk, 0, stream>>>(Q1, K2, Sc);

    row_stats_kernel<<<dim3(BATCH * SEQ / 8), blk, 0, stream>>>(Sc, Mx, Rz);
    colsum_kernel<<<dim3(SEQ / 256, BATCH), blk, 0, stream>>>(Sc, Mx, Rz, Wv);
    out_kernel<<<dim3(D_DIM / 256, BATCH), blk, 0, stream>>>(Wv, V2, out);
}
